// FlatLstm_3719441678851
// MI455X (gfx1250) — matrix operand order repaired and checked by reading, not compiled, not run on hardware
//
#include <hip/hip_runtime.h>

// N=1024, S=8, I=128, F=1024, M=512.
// h0 = c0 = 0  =>  Wh* and the forget gate are dead; only Wxi/Wxg/Wxo + Wout matter.

typedef __attribute__((ext_vector_type(16))) __bf16 v16bf;
typedef __attribute__((ext_vector_type(8)))  float  v8f;
typedef __attribute__((ext_vector_type(4)))  float  v4f;
typedef __attribute__((ext_vector_type(4)))  __bf16 v4bf;

#define N_TOK 1024
#define S_STR 8
#define I_DIM 128
#define F_DIM 1024
#define M_DIM 512

// LDS row: 128 bf16 + 8 pad = 136 (272B stride -> conflict-free b128 reads)
#define LDSW 136

// ---------------------------------------------------------------------------
// helpers
// ---------------------------------------------------------------------------
__device__ __forceinline__ float tanh_hw(float x) {
#if __has_builtin(__builtin_amdgcn_tanhf)
  return __builtin_amdgcn_tanhf(x);          // v_tanh_f32 (gfx1250 TRANS op)
#else
  x = fminf(fmaxf(x, -15.0f), 15.0f);
  float e = __expf(2.0f * x);
  return (e - 1.0f) / (e + 1.0f);
#endif
}
__device__ __forceinline__ float sigm_hw(float x) {
  return __builtin_fmaf(0.5f, tanh_hw(0.5f * x), 0.5f);   // sigma(x)=0.5*tanh(x/2)+0.5
}

__device__ __forceinline__ unsigned short bf16_bits(float x) {
  __bf16 b = (__bf16)x;                       // native cvt, RNE
  return __builtin_bit_cast(unsigned short, b);
}

// Per-lane async copy of 16B global -> LDS (VGLOBAL GLOBAL_LOAD_ASYNC_TO_LDS_B128).
__device__ __forceinline__ void async_copy_b128(const unsigned short* gsrc,
                                                unsigned short* lds_dst) {
  unsigned lofs = (unsigned)(uintptr_t)lds_dst;            // low 32 bits = LDS offset
  unsigned long long ga = (unsigned long long)(uintptr_t)gsrc;
  asm volatile("global_load_async_to_lds_b128 %0, %1, off"
               :: "v"(lofs), "v"(ga) : "memory");
}
__device__ __forceinline__ void wait_async0() {
  asm volatile("s_wait_asynccnt 0x0" ::: "memory");
}

union Frag {
  v16bf v;
  uint4 q[2];
};

// ---------------------------------------------------------------------------
// Kernel 0: streaming fp32 -> bf16 conversion (one pass, grid-stride)
// ---------------------------------------------------------------------------
__global__ __launch_bounds__(256) void cvt_bf16_kernel(const float* __restrict__ src,
                                                       unsigned short* __restrict__ dst,
                                                       int n4) {
  int idx    = blockIdx.x * 256 + threadIdx.x;
  int stride = gridDim.x * 256;
  for (int i = idx; i < n4; i += stride) {
    v4f x = ((const v4f*)src)[i];
    v4bf b = __builtin_convertvector(x, v4bf);             // v_cvt_pk_bf16_f32 x2
    ((ulonglong1*)dst)[i] = __builtin_bit_cast(ulonglong1, b);
  }
}

// ---------------------------------------------------------------------------
// Kernel A: gate GEMMs (K=I=128) + LSTM elementwise -> h (bf16) workspace
// Block 256 thr (8 waves); tile 128n x 64f, one stream. All operands bf16.
// ---------------------------------------------------------------------------
__global__ __launch_bounds__(256) void lstm_gates_kernel(
    const unsigned short* __restrict__ xb,    // bf16 [N][S][I]
    const unsigned short* __restrict__ Wib,   // bf16 [S][F][I]
    const unsigned short* __restrict__ Wgb,
    const unsigned short* __restrict__ Wob,
    const float* __restrict__ bi, const float* __restrict__ bg,
    const float* __restrict__ bo,
    unsigned short* __restrict__ h_ws) {
  __shared__ unsigned short sW[3][64 * LDSW];

  const int s     = blockIdx.z;
  const int f_blk = blockIdx.y * 64;
  const int n_blk = blockIdx.x * 128;
  const int tid   = threadIdx.x;
  const int wave  = tid >> 5;
  const int lane  = tid & 31;
  const int half  = lane >> 4;
  const int l16   = lane & 15;

  const unsigned short* Wg3[3] = {
    Wib + (size_t)s * F_DIM * I_DIM + (size_t)f_blk * I_DIM,
    Wgb + (size_t)s * F_DIM * I_DIM + (size_t)f_blk * I_DIM,
    Wob + (size_t)s * F_DIM * I_DIM + (size_t)f_blk * I_DIM };

  // Async fill: per gate 64 rows x 128 bf16 = 1024 x 16B chunks; 4 per thread.
  for (int g = 0; g < 3; ++g) {
    for (int j = 0; j < 4; ++j) {
      int e  = tid + 256 * j;          // chunk id 0..1023
      int fl = e >> 4;                 // row 0..63
      int c  = e & 15;                 // 8-ushort chunk in row
      async_copy_b128(Wg3[g] + (size_t)fl * I_DIM + c * 8,
                      &sW[g][fl * LDSW + c * 8]);
    }
  }
  wait_async0();
  __syncthreads();

  const unsigned short* xrow =
      xb + (size_t)(n_blk + wave * 16 + l16) * (S_STR * I_DIM) + s * I_DIM;

  v8f acc[3][4] = {};

  for (int kk = 0; kk < 4; ++kk) {     // K = 128 = 4 x 32
    Frag a;                            // bf16 A frag straight from global
    const unsigned short* ap = xrow + kk * 32 + half * 8;
    a.q[0] = ((const uint4*)ap)[0];
    a.q[1] = ((const uint4*)(ap + 16))[0];
    for (int g = 0; g < 3; ++g) {
      for (int t = 0; t < 4; ++t) {
        Frag b;
        const unsigned short* bp = &sW[g][(t * 16 + l16) * LDSW + kk * 32 + half * 8];
        b.q[0] = ((const uint4*)bp)[0];
        b.q[1] = ((const uint4*)(bp + 16))[0];
        acc[g][t] = __builtin_amdgcn_wmma_f32_16x16x32_bf16(
            false, a.v, false, b.v, (short)0, acc[g][t], false, false);
      }
    }
  }

  // c = sig(pi)*tanh(pg); h = sig(po)*tanh(c)   (all via v_tanh_f32)
  for (int t = 0; t < 4; ++t) {
    int f = f_blk + t * 16 + l16;
    float vbi = bi[s * F_DIM + f];
    float vbg = bg[s * F_DIM + f];
    float vbo = bo[s * F_DIM + f];
    for (int r = 0; r < 8; ++r) {
      int rr = n_blk + wave * 16 + r + half * 8;       // C layout: M = r + 8*half
      float c = sigm_hw(acc[0][t][r] + vbi) * tanh_hw(acc[1][t][r] + vbg);
      float h = sigm_hw(acc[2][t][r] + vbo) * tanh_hw(c);
      h_ws[(size_t)rr * (S_STR * F_DIM) + s * F_DIM + f] = bf16_bits(h);
    }
  }
}

// ---------------------------------------------------------------------------
// Kernel B: out = h @ Wout^T + bout  (K=F=1024, 8 LDS-staged chunks of 128)
// ---------------------------------------------------------------------------
__global__ __launch_bounds__(256) void lstm_out_kernel(
    const unsigned short* __restrict__ h_ws,   // bf16 [N][S][F]
    const unsigned short* __restrict__ Wb,     // bf16 [S][M][F]
    const float* __restrict__ bout,
    float* __restrict__ out) {
  __shared__ unsigned short sB[64 * LDSW];

  const int s     = blockIdx.z;
  const int m_blk = blockIdx.y * 64;
  const int n_blk = blockIdx.x * 128;
  const int tid   = threadIdx.x;
  const int wave  = tid >> 5;
  const int lane  = tid & 31;
  const int half  = lane >> 4;
  const int l16   = lane & 15;

  const unsigned short* Wbase = Wb + (size_t)s * M_DIM * F_DIM + (size_t)m_blk * F_DIM;
  const unsigned short* arow =
      h_ws + (size_t)(n_blk + wave * 16 + l16) * (S_STR * F_DIM) + s * F_DIM;

  v8f acc[4] = {};

  for (int kc = 0; kc < 8; ++kc) {
    // Async fill: 64 m-rows x 128 bf16 chunk = 1024 x 16B; 4 per thread.
    for (int j = 0; j < 4; ++j) {
      int e  = tid + 256 * j;
      int fl = e >> 4;
      int c  = e & 15;
      async_copy_b128(Wbase + (size_t)fl * F_DIM + kc * 128 + c * 8,
                      &sB[fl * LDSW + c * 8]);
    }
    wait_async0();
    __syncthreads();

    // Prefetch next chunk's cachelines (128 rows x 2 x 128B covers 128x128 bf16, 64 rows past the tile).
    if (kc < 7)
      __builtin_prefetch(Wbase + (size_t)(tid >> 1) * F_DIM + (kc + 1) * 128 + (tid & 1) * 64, 0, 1);

    for (int kk = 0; kk < 4; ++kk) {
      Frag a;
      const unsigned short* ap = arow + kc * 128 + kk * 32 + half * 8;
      a.q[0] = ((const uint4*)ap)[0];
      a.q[1] = ((const uint4*)(ap + 16))[0];
      for (int t = 0; t < 4; ++t) {
        Frag b;
        const unsigned short* bp = &sB[(t * 16 + l16) * LDSW + kk * 32 + half * 8];
        b.q[0] = ((const uint4*)bp)[0];
        b.q[1] = ((const uint4*)(bp + 16))[0];
        acc[t] = __builtin_amdgcn_wmma_f32_16x16x32_bf16(
            false, a.v, false, b.v, (short)0, acc[t], false, false);
      }
    }
    __syncthreads();
  }

  for (int t = 0; t < 4; ++t) {
    int m = m_blk + t * 16 + l16;
    float vb = bout[s * M_DIM + m];
    for (int r = 0; r < 8; ++r) {
      int rr = n_blk + wave * 16 + r + half * 8;
      out[(size_t)rr * (S_STR * M_DIM) + s * M_DIM + m] = acc[t][r] + vb;
    }
  }
}

// ---------------------------------------------------------------------------
// Inputs: 0 modulation, 1 Wxi, 2 Whi, 3 bi, 4 Wxf, 5 Whf, 6 bf, 7 Wxg, 8 Whg,
// 9 bg, 10 Wxo, 11 Who, 12 bo, 13 Wout, 14 bout.  Wh*/Wxf dead (h0=c0=0).
// Workspace (ushort units): h[8M] | xb[1M] | Wib[1M] | Wgb[1M] | Wob[1M] | Woutb[4M]
// ---------------------------------------------------------------------------
extern "C" void kernel_launch(void* const* d_in, const int* in_sizes, int n_in,
                              void* d_out, int out_size, void* d_ws, size_t ws_size,
                              hipStream_t stream) {
  (void)in_sizes; (void)n_in; (void)out_size; (void)ws_size;
  const float* mod  = (const float*)d_in[0];
  const float* Wxi  = (const float*)d_in[1];
  const float* bi   = (const float*)d_in[3];
  const float* Wxg  = (const float*)d_in[7];
  const float* bg   = (const float*)d_in[9];
  const float* Wxo  = (const float*)d_in[10];
  const float* bo   = (const float*)d_in[12];
  const float* Wout = (const float*)d_in[13];
  const float* bout = (const float*)d_in[14];

  unsigned short* ws = (unsigned short*)d_ws;
  unsigned short* h_ws  = ws;                                  // 8,388,608
  unsigned short* xb    = ws + 8388608;                        // 1,048,576
  unsigned short* Wib   = ws + 8388608 + 1048576;              // 1,048,576
  unsigned short* Wgb   = ws + 8388608 + 2 * 1048576;          // 1,048,576
  unsigned short* Wob   = ws + 8388608 + 3 * 1048576;          // 1,048,576
  unsigned short* Woutb = ws + 8388608 + 4 * 1048576;          // 4,194,304

  // One-time fp32 -> bf16 packing (streaming, ~34 MB read).
  cvt_bf16_kernel<<<256, 256, 0, stream>>>(mod,  xb,    (N_TOK * S_STR * I_DIM) / 4);
  cvt_bf16_kernel<<<256, 256, 0, stream>>>(Wxi,  Wib,   (S_STR * F_DIM * I_DIM) / 4);
  cvt_bf16_kernel<<<256, 256, 0, stream>>>(Wxg,  Wgb,   (S_STR * F_DIM * I_DIM) / 4);
  cvt_bf16_kernel<<<256, 256, 0, stream>>>(Wxo,  Wob,   (S_STR * F_DIM * I_DIM) / 4);
  cvt_bf16_kernel<<<256, 256, 0, stream>>>(Wout, Woutb, (S_STR * M_DIM * F_DIM) / 4);

  dim3 gridA(N_TOK / 128, F_DIM / 64, S_STR);    // 8 x 16 x 8
  lstm_gates_kernel<<<gridA, 256, 0, stream>>>(xb, Wib, Wgb, Wob, bi, bg, bo, h_ws);

  dim3 gridB(N_TOK / 128, M_DIM / 64, S_STR);    // 8 x 8 x 8
  lstm_out_kernel<<<gridB, 256, 0, stream>>>(h_ws, Woutb, bout, (float*)d_out);
}
